// GCN_33603824124473
// MI455X (gfx1250) — matrix operand order repaired and checked by reading, not compiled, not run on hardware
//
#include <hip/hip_runtime.h>
#include <hip/hip_bf16.h>

// ---------------------------------------------------------------------------
// 2-layer GCN for MI455X (gfx1250, wave32).
//  GEMMs: bf16 WMMA (v_wmma_f32_16x16x32_bf16) with double-buffered LDS tiles
//         filled by async global->LDS DMA (global_load_async_to_lds_b128,
//         ASYNCcnt-tracked) so tile k+1 streams in while tile k is computed.
//         K-loop manually unrolled 2x so the ping-pong buffer index is a
//         compile-time constant (no cndmask/hazard-NOPs in the hot loop).
//  Aggregation: edge-parallel float4 gather + global_atomic_add_f32 scatter.
//  Softmax: wave32 shuffle reductions.
// ---------------------------------------------------------------------------

typedef __attribute__((ext_vector_type(16))) __bf16 v16bf;
typedef __attribute__((ext_vector_type(8)))  __bf16 v8bf;
typedef __attribute__((ext_vector_type(4)))  __bf16 v4bf;
typedef __attribute__((ext_vector_type(8)))  float  v8f;

// Async 16B global->LDS copy (per-lane). GVS addressing: uniform 64-bit SGPR
// base + per-lane 32-bit byte offset. vdst VGPR holds the LDS byte address
// (low 32 bits of the generic shared pointer per flat-address truncation).
__device__ __forceinline__ void async_copy16(unsigned lds_addr,
                                             const void* sbase,
                                             unsigned byte_off)
{
    asm volatile("global_load_async_to_lds_b128 %0, %1, %2"
                 :: "v"(lds_addr), "v"(byte_off), "s"(sbase)
                 : "memory");
}

__device__ __forceinline__ void wait_async0()
{
    asm volatile("s_wait_asynccnt 0x0" ::: "memory");
}

// ---------------------------------------------------------------------------
// GEMM: C[M][N] = A[M][K] (bf16, row-major) * B (given as BT[N][K] bf16)
// BM x BN block tile, BK=32 K-step, 8 waves of 32 lanes.
// Per-wave tile = (TM*16) x (TN*16).  Requires (K / 32) even.
// ---------------------------------------------------------------------------
template<int BM, int BN, int TM, int TN>
__global__ __launch_bounds__(256) void gemm_bf16_wmma(
    const __bf16* __restrict__ A, const __bf16* __restrict__ BT,
    float* __restrict__ C, int M, int N, int K)
{
    constexpr int BK = 32;
    constexpr int WROWS = BM / (TM * 16);
    constexpr int WCOLS = BN / (TN * 16);
    static_assert(WROWS * WCOLS == 8, "must use exactly 8 wave32s");
    constexpr int LDSP = BK + 8;   // +8 bf16 pad keeps 16B alignment, shifts banks

    __shared__ __bf16 As[2][BM][LDSP];
    __shared__ __bf16 Bs[2][BN][LDSP];

    const int tid  = threadIdx.x;
    const int wave = tid >> 5;
    const int lane = tid & 31;
    const int half = lane >> 4;     // which 16-lane half of the wave
    const int l16  = lane & 15;
    const int wrow = wave % WROWS;
    const int wcol = wave / WROWS;
    const int rowBase = blockIdx.x * BM;
    const int colBase = blockIdx.y * BN;

    v8f acc[TM][TN] = {};

    // LDS fill mapping: thread -> (row, 16-bf16 segment); 16B per thread/matrix
    const int fr = tid >> 1;
    const int fs = tid & 1;

    const unsigned aLds[2] = { (unsigned)(uintptr_t)&As[0][fr][fs * 16],
                               (unsigned)(uintptr_t)&As[1][fr][fs * 16] };
    const unsigned bLds[2] = { (unsigned)(uintptr_t)&Bs[0][fr][fs * 16],
                               (unsigned)(uintptr_t)&Bs[1][fr][fs * 16] };
    const unsigned aOffBase = (unsigned)(((size_t)(rowBase + fr) * K + fs * 16) * sizeof(__bf16));
    const unsigned bOffBase = (unsigned)(((size_t)(colBase + fr) * K + fs * 16) * sizeof(__bf16));

    // Prologue: start DMA of tile 0 into buffer 0.
    async_copy16(aLds[0], A, aOffBase);
    if (fr < BN) async_copy16(bLds[0], BT, bOffBase);

    for (int kb = 0; kb < K; kb += 2 * BK) {
        #pragma unroll
        for (int u = 0; u < 2; ++u) {           // u: compile-time buffer index
            const int cur = u;
            const int nxt = u ^ 1;
            const int kk  = kb + u * BK;

            wait_async0();          // my async writes for tile kk have landed
            __syncthreads();        // everyone's have landed

            // Kick off DMA of the next tile into the other buffer; it
            // overlaps with the fragment loads + WMMAs below.
            if (kk + BK < K) {
                unsigned koff = (unsigned)((kk + BK) * sizeof(__bf16));
                async_copy16(aLds[nxt], A, aOffBase + koff);
                if (fr < BN) async_copy16(bLds[nxt], BT, bOffBase + koff);
            }

            // Fragment loads per ISA §7.12.2 layouts (16-bit A 16x32, B 32x16).
            union Frag { v16bf v; v8bf h[2]; } af[TM], bf[TN];
            #pragma unroll
            for (int i = 0; i < TM; ++i) {
                const __bf16* ap = &As[cur][wrow * TM * 16 + i * 16 + l16][0];
                af[i].h[0] = *(const v8bf*)(ap + half * 8);      // VGPR0-3
                af[i].h[1] = *(const v8bf*)(ap + 16 + half * 8); // VGPR4-7
            }
            #pragma unroll
            for (int j = 0; j < TN; ++j) {
                const __bf16* bp = &Bs[cur][wcol * TN * 16 + j * 16 + l16][0];
                bf[j].h[0] = *(const v8bf*)(bp + half * 8);      // k lo
                bf[j].h[1] = *(const v8bf*)(bp + 16 + half * 8); // k hi
            }

            #pragma unroll
            for (int i = 0; i < TM; ++i)
                #pragma unroll
                for (int j = 0; j < TN; ++j)
                    acc[i][j] = __builtin_amdgcn_wmma_f32_16x16x32_bf16(
                        false, af[i].v, false, bf[j].v,
                        (short)0, acc[i][j], false, false);

            // All shared reads of buffer `cur` retire before this barrier
            // signal (s_wait_dscnt precedes the WMMAs), so the following
            // iteration may safely DMA into it.
            __syncthreads();
        }
    }

    // D layout: VGPR r holds (M = r + 8*half, N = l16) of each 16x16 tile.
    #pragma unroll
    for (int i = 0; i < TM; ++i) {
        #pragma unroll
        for (int j = 0; j < TN; ++j) {
            const int col = colBase + wcol * TN * 16 + j * 16 + l16;
            float* cp = C + (size_t)(rowBase + wrow * TM * 16 + i * 16 + half * 8) * N + col;
            #pragma unroll
            for (int rr = 0; rr < 8; ++rr)
                cp[(size_t)rr * N] = acc[i][j][rr];
        }
    }
}

// ---------------------------------------------------------------------------
// fp32 -> bf16 cast (optionally fused ReLU), 4 elements / thread
// ---------------------------------------------------------------------------
template<bool RELU>
__global__ __launch_bounds__(256) void cast_bf16_kernel(
    const float* __restrict__ in, __bf16* __restrict__ out, size_t n4)
{
    size_t i = (size_t)blockIdx.x * blockDim.x + threadIdx.x;
    if (i >= n4) return;
    float4 v = ((const float4*)in)[i];
    if (RELU) {
        v.x = fmaxf(v.x, 0.f); v.y = fmaxf(v.y, 0.f);
        v.z = fmaxf(v.z, 0.f); v.w = fmaxf(v.w, 0.f);
    }
    v4bf o;
    o.x = (__bf16)v.x; o.y = (__bf16)v.y; o.z = (__bf16)v.z; o.w = (__bf16)v.w;
    ((v4bf*)out)[i] = o;
}

// W[K][N] fp32 -> WT[N][K] bf16 (tiny, once per layer)
__global__ __launch_bounds__(256) void transpose_cast_kernel(
    const float* __restrict__ W, __bf16* __restrict__ WT, int K, int N)
{
    int i = blockIdx.x * blockDim.x + threadIdx.x;
    if (i >= K * N) return;
    int k = i / N, n = i % N;
    WT[(size_t)n * K + k] = (__bf16)W[(size_t)k * N + n];
}

// ---------------------------------------------------------------------------
// Degree / normalization
// ---------------------------------------------------------------------------
__global__ __launch_bounds__(256) void deg_init_kernel(float* deg, int n)
{
    int i = blockIdx.x * blockDim.x + threadIdx.x;
    if (i < n) deg[i] = 1.0f;                       // self-loop
}

__global__ __launch_bounds__(256) void deg_accum_kernel(
    const long long* __restrict__ dst, float* deg, int E)
{
    int e = blockIdx.x * blockDim.x + threadIdx.x;
    if (e < E) atomicAdd(&deg[dst[e]], 1.0f);
}

__global__ __launch_bounds__(256) void dinv_kernel(float* deg, int n)
{
    int i = blockIdx.x * blockDim.x + threadIdx.x;
    if (i < n) {
        float d = deg[i];
        deg[i] = (d > 0.f) ? rsqrtf(d) : 0.f;       // in-place deg -> dinv
    }
}

// H[i][:] = bias + XW[i][:] * dinv[i]^2   (self-loop term + atomic-buffer init)
__global__ __launch_bounds__(256) void self_init_kernel(
    const float* __restrict__ XW, const float* __restrict__ dinv,
    const float* __restrict__ bias, float* __restrict__ H, int N, int D)
{
    size_t i = (size_t)blockIdx.x * blockDim.x + threadIdx.x;
    size_t total = (size_t)N * (D / 4);
    if (i >= total) return;
    int node = (int)(i / (D / 4));
    int c    = (int)(i % (D / 4)) * 4;
    float s = dinv[node]; s *= s;
    float4 v = *(const float4*)(XW + (size_t)node * D + c);
    float4 b = *(const float4*)(bias + c);
    float4 o = { v.x * s + b.x, v.y * s + b.y, v.z * s + b.z, v.w * s + b.w };
    *(float4*)(H + (size_t)node * D + c) = o;
}

// H[dst][:] += XW[src][:] * dinv[src]*dinv[dst]   (D/4 float4 chunks per edge)
template<int D>
__global__ __launch_bounds__(256) void edge_agg_kernel(
    const long long* __restrict__ ei, const float* __restrict__ XW,
    const float* __restrict__ dinv, float* __restrict__ H, int E)
{
    constexpr int CH = D / 4;
    size_t tid = (size_t)blockIdx.x * blockDim.x + threadIdx.x;
    if (tid >= (size_t)E * CH) return;
    int e = (int)(tid / CH);
    int c = (int)(tid % CH) * 4;
    long long s = ei[e];
    long long d = ei[(size_t)E + e];
    float nrm = dinv[s] * dinv[d];
    float4 v = *(const float4*)(XW + (size_t)s * D + c);
    float* out = H + (size_t)d * D + c;
    atomicAdd(out + 0, v.x * nrm);
    atomicAdd(out + 1, v.y * nrm);
    atomicAdd(out + 2, v.z * nrm);
    atomicAdd(out + 3, v.w * nrm);
}

// log_softmax over rows of 64; one wave32 per row, 2 values per lane
__global__ __launch_bounds__(256) void log_softmax64_kernel(
    const float* __restrict__ O, float* __restrict__ out, int N)
{
    int wave = threadIdx.x >> 5;
    int lane = threadIdx.x & 31;
    int row  = blockIdx.x * (blockDim.x >> 5) + wave;
    if (row >= N) return;
    const float* p = O + (size_t)row * 64;
    float a = p[lane], b = p[lane + 32];
    float m = fmaxf(a, b);
    #pragma unroll
    for (int off = 16; off; off >>= 1) m = fmaxf(m, __shfl_xor(m, off, 32));
    float s = __expf(a - m) + __expf(b - m);
    #pragma unroll
    for (int off = 16; off; off >>= 1) s += __shfl_xor(s, off, 32);
    float ls = __logf(s);
    float* q = out + (size_t)row * 64;
    q[lane]      = a - m - ls;
    q[lane + 32] = b - m - ls;
}

// ---------------------------------------------------------------------------
extern "C" void kernel_launch(void* const* d_in, const int* in_sizes, int n_in,
                              void* d_out, int out_size, void* d_ws, size_t ws_size,
                              hipStream_t stream)
{
    const float*     x   = (const float*)d_in[0];   // [65536][512]
    const float*     W1  = (const float*)d_in[1];   // [512][256]
    const float*     b1  = (const float*)d_in[2];   // [256]
    const float*     W2  = (const float*)d_in[3];   // [256][64]
    const float*     b2  = (const float*)d_in[4];   // [64]
    const long long* ei  = (const long long*)d_in[5]; // [2][1048576] int64

    constexpr int N   = 65536;
    constexpr int E   = 1048576;
    constexpr int DIN = 512, DHID = 256, DOUT = 64;

    char* ws = (char*)d_ws;
    __bf16* Xb   = (__bf16*)(ws);                                  // 64 MB
    float*  XW1  = (float*)(ws + (size_t)(64u  << 20));            // 64 MB
    float*  H    = (float*)(ws + (size_t)(128u << 20));            // 64 MB
    float*  dinv = (float*)(ws + (size_t)(192u << 20));            // 256 KB
    __bf16* W1T  = (__bf16*)(ws + (size_t)(192u << 20) + (1u << 18)); // 256 KB
    __bf16* W2T  = (__bf16*)(ws + (size_t)(192u << 20) + (2u << 18)); // 32 KB
    __bf16* Hb   = Xb;    // Xb dead after GEMM1
    float*  HW2  = XW1;   // XW1 dead after layer-1 aggregation
    float*  O    = H;     // H dead after relu-cast

    // --- precompute: casts, transposed bf16 weights, degree normalization ---
    {
        size_t n4 = (size_t)N * DIN / 4;
        cast_bf16_kernel<false><<<(int)((n4 + 255) / 256), 256, 0, stream>>>(x, Xb, n4);
    }
    transpose_cast_kernel<<<(DIN * DHID + 255) / 256, 256, 0, stream>>>(W1, W1T, DIN, DHID);
    transpose_cast_kernel<<<(DHID * DOUT + 255) / 256, 256, 0, stream>>>(W2, W2T, DHID, DOUT);
    deg_init_kernel<<<N / 256, 256, 0, stream>>>(dinv, N);
    deg_accum_kernel<<<E / 256, 256, 0, stream>>>(ei + E, dinv, E);  // dst row
    dinv_kernel<<<N / 256, 256, 0, stream>>>(dinv, N);

    // --- layer 1: XW1 = X @ W1 (bf16 WMMA), then normalized scatter-add ---
    gemm_bf16_wmma<128, 128, 2, 4><<<dim3(N / 128, DHID / 128), 256, 0, stream>>>(
        Xb, W1T, XW1, N, DHID, DIN);
    self_init_kernel<<<(int)((size_t)N * (DHID / 4) / 256), 256, 0, stream>>>(
        XW1, dinv, b1, H, N, DHID);
    edge_agg_kernel<DHID><<<(int)((size_t)E * (DHID / 4) / 256), 256, 0, stream>>>(
        ei, XW1, dinv, H, E);
    {
        size_t n4 = (size_t)N * DHID / 4;
        cast_bf16_kernel<true><<<(int)((n4 + 255) / 256), 256, 0, stream>>>(H, Hb, n4);
    }

    // --- layer 2: HW2 = relu(H) @ W2, scatter-add, log_softmax ---
    gemm_bf16_wmma<128, 64, 2, 2><<<dim3(N / 128, DOUT / 64), 256, 0, stream>>>(
        Hb, W2T, HW2, N, DOUT, DHID);
    self_init_kernel<<<(int)((size_t)N * (DOUT / 4) / 256), 256, 0, stream>>>(
        HW2, dinv, b2, O, N, DOUT);
    edge_agg_kernel<DOUT><<<(int)((size_t)E * (DOUT / 4) / 256), 256, 0, stream>>>(
        ei, HW2, dinv, O, E);
    log_softmax64_kernel<<<N / 8, 256, 0, stream>>>(O, (float*)d_out, N);
}
